// RetinaNet_1005022347407
// MI455X (gfx1250) — matrix operand order repaired and checked by reading, not compiled, not run on hardware
//
#include <hip/hip_runtime.h>
#include <hip/hip_bf16.h>

// ---------------------------------------------------------------------------
// RetinaNet head on MI455X (gfx1250, wave32) via implicit-GEMM WMMA f16.
//
// Compute-bound (~0.35 TFLOP vs ~110MB HBM traffic) -> all conv math goes
// through v_wmma_f32_16x16x32_f16. Activations kept as zero-padded NHWC f16
// in workspace; weights repacked once to f16 [O][9*256] A-matrices.
// B tiles staged into LDS with gfx1250 async copies
// (global_load_async_to_lds_b128, ASYNCcnt) in a double-buffered pipeline:
// phase p+1's async loads are in flight while phase p's WMMAs execute.
// All kernels templated on (H,W) so index math folds to shifts/mults.
// ---------------------------------------------------------------------------

typedef __attribute__((ext_vector_type(16))) _Float16 v16h;
typedef __attribute__((ext_vector_type(8)))  float    v8f;

#define CH    256
#define KDIM  2304           // 9 * 256
#define AANCH 9
#define KCLS  80
#define RTOT  120087

#define PH_STEPS 24          // K-steps (32 K each) per pipeline phase; 3 phases
#define PH_HALFS (PH_STEPS * 512)   // 12288 halfs = 24KB per LDS buffer

// ---------------- weight repack: (O,C,3,3) f32 -> [Opad][9*256] f16 --------
__global__ __launch_bounds__(256) void pack_w(const float* __restrict__ w,
                                              _Float16* __restrict__ out,
                                              int O, int Opad) {
  int idx = blockIdx.x * 256 + threadIdx.x;
  int total = Opad * KDIM;
  if (idx >= total) return;
  int o = idx / KDIM;
  int k = idx - o * KDIM;
  int t = k >> 8;            // tap 0..8
  int c = k & 255;
  float v = 0.0f;
  if (o < O) v = w[((o * CH + c) * 3 + (t / 3)) * 3 + (t % 3)];
  out[idx] = (_Float16)v;
}

// ---------------- feature: (N,C,H,W) f32 -> padded NHWC f16 ----------------
template <int H, int W>
__global__ __launch_bounds__(256) void to_padded(const float* __restrict__ x,
                                                 _Float16* __restrict__ o) {
  constexpr int Wp = W + 2, Hp = H + 2;
  constexpr int total = 2 * Hp * Wp * CH;
  int idx = blockIdx.x * 256 + threadIdx.x;
  if (idx >= total) return;
  int c = idx & 255;
  int r = idx >> 8;
  int pw = r % Wp; r /= Wp;
  int ph = r % Hp; int nn = r / Hp;
  float v = 0.0f;
  if (ph >= 1 && ph <= H && pw >= 1 && pw <= W)
    v = x[((nn * CH + c) * H + (ph - 1)) * W + (pw - 1)];
  o[idx] = (_Float16)v;
}

__global__ __launch_bounds__(256) void zero_f16(_Float16* __restrict__ p, int n8) {
  int idx = blockIdx.x * 256 + threadIdx.x;
  if (idx < n8) {
    uint4 z; z.x = z.y = z.z = z.w = 0u;
    ((uint4*)p)[idx] = z;
  }
}

// ---------------- shared GEMM core helpers ---------------------------------
struct PixInfo {
  int n, p;
  int pixBase;   // padded NHWC base offset of tap (dh=0,dw=0)
  bool valid;
};

template <int H, int W>
__device__ __forceinline__ PixInfo pix_of(int g0, int lane15) {
  constexpr int HW = H * W;
  constexpr int total = 2 * HW;
  constexpr int Wp = W + 2, Hp = H + 2;
  PixInfo pi;
  int g = g0 + lane15;
  pi.valid = (g < total);
  if (g >= total) g = total - 1;
  pi.n = g / HW;
  pi.p = g - pi.n * HW;
  int h = pi.p / W;
  int w = pi.p - h * W;
  pi.pixBase = ((pi.n * Hp + h) * Wp + w) * CH;
  (void)Hp;
  return pi;
}

__device__ __forceinline__ void wait_async_all() {
  asm volatile("s_wait_asynccnt 0x0" ::: "memory");
}

// Issue one LDS phase (PH_STEPS K-steps x 16 px x 32 K of B) as gfx1250
// async copies; caller drains with s_wait_asynccnt + barrier.
// Thread mapping: col = tid&15 (one pixel decomposition per thread),
// tid>>4 picks 6 of the 96 (step, k-quarter) pairs.
template <int H, int W>
__device__ __forceinline__ void fill_issue(unsigned ldsBase,
                                           const _Float16* __restrict__ act,
                                           int ph, int blockPix0) {
  constexpr int Wp = W + 2;
  const int tid = threadIdx.x;
  const int col = tid & 15;
  const int idx16 = tid >> 4;
  PixInfo pi = pix_of<H, W>(blockPix0, col);
  const _Float16* base = act + pi.pixBase;
#pragma unroll
  for (int i = 0; i < 6; ++i) {
    int pr = idx16 + 16 * i;          // 0..95
    int s = pr >> 2;                  // step in phase 0..23
    int kq = (pr & 3) << 3;           // k offset 0/8/16/24 (halfs)
    int sG = ph * PH_STEPS + s;       // global K-step 0..71
    int t = sG >> 3, kb = sG & 7;     // tap, 32-k block
    int dh = t / 3, dw = t - dh * 3;
    const void* src = (const void*)(base + (dh * Wp + dw) * CH + kb * 32 + kq);
    unsigned lds = ldsBase + (unsigned)(s * 1024 + col * 64 + (kq << 1));
    asm volatile("global_load_async_to_lds_b128 %0, %1, off"
                 :: "v"(lds), "v"(src) : "memory");
  }
}

// GEMM inner: LDS B + packed-weight A; A K-offset is contiguous (32*sG).
template <int MT>
__device__ __forceinline__ void gemm_phase(v8f* acc, const _Float16* ldsB,
                                           const _Float16* __restrict__ wgt,
                                           int ph, int wave, int col, int kSel,
                                           int Mtiles) {
  const _Float16* bp0 = ldsB + col * 32 + kSel * 16;
  const _Float16* ap0 = wgt + col * KDIM + ph * (PH_STEPS * 32) + kSel * 16;
#pragma unroll 12
  for (int s = 0; s < PH_STEPS; ++s) {
    v16h bfrag;
    const _Float16* bp = bp0 + s * 512;
    *(uint4*)&bfrag       = *(const uint4*)bp;
    *((uint4*)&bfrag + 1) = *(const uint4*)(bp + 8);
#pragma unroll
    for (int j = 0; j < MT; ++j) {
      int mTile = wave + 8 * j;                    // wave-uniform
      if (mTile < Mtiles) {
        const _Float16* ap = ap0 + mTile * (16 * KDIM) + s * 32;
        v16h afrag;
        *(uint4*)&afrag       = *(const uint4*)ap;
        *((uint4*)&afrag + 1) = *(const uint4*)(ap + 8);
        acc[j] = __builtin_amdgcn_wmma_f32_16x16x32_f16(
            false, afrag, false, bfrag, (short)0, acc[j], false, false);
      }
    }
  }
}

// Double-buffered async pipeline over 3 phases of PH_STEPS each.
template <int MT, int H, int W>
__device__ __forceinline__ void gemm_pipeline(v8f* acc, _Float16 (*ldsB)[PH_HALFS],
                                              const _Float16* __restrict__ act,
                                              const _Float16* __restrict__ wgt,
                                              int blockPix0, int wave, int col,
                                              int kSel, int Mtiles) {
  const unsigned base0 = (unsigned)(uintptr_t)(&ldsB[0][0]);
  const unsigned base1 = (unsigned)(uintptr_t)(&ldsB[1][0]);
  fill_issue<H, W>(base0, act, 0, blockPix0);
  wait_async_all();
  __syncthreads();
#pragma unroll
  for (int ph = 0; ph < 3; ++ph) {
    if (ph < 2)                       // next phase's copies fly under the WMMAs
      fill_issue<H, W>((ph & 1) ? base0 : base1, act, ph + 1, blockPix0);
    gemm_phase<MT>(acc, &ldsB[ph & 1][0], wgt, ph, wave, col, kSel, Mtiles);
    if (ph < 2) {
      wait_async_all();
      __syncthreads();                // also fences buffer reuse at ph+2
    }
  }
}

// ---------------- tower conv: GEMM + bias + ReLU -> padded NHWC f16 --------
template <int MT, int H, int W>
__global__ __launch_bounds__(256) void conv_wmma(
    const _Float16* __restrict__ act, const _Float16* __restrict__ wgt,
    const float* __restrict__ bias, _Float16* __restrict__ outAct,
    int Mtiles) {
  __shared__ _Float16 ldsB[2][PH_HALFS];    // 2 x 24KB
  constexpr int Wp = W + 2;
  const int lane = threadIdx.x & 31;
  const int wave = threadIdx.x >> 5;
  const int col  = lane & 15;
  const int kSel = lane >> 4;               // 0: K 0..15 / M 0..7, 1: K 16..31 / M 8..15
  const int blockPix0 = blockIdx.x * 16;

  PixInfo pi = pix_of<H, W>(blockPix0, col);

  v8f zero = {};
  v8f acc[MT];
#pragma unroll
  for (int j = 0; j < MT; ++j) acc[j] = zero;

  gemm_pipeline<MT, H, W>(acc, ldsB, act, wgt, blockPix0, wave, col, kSel, Mtiles);

  const int outBase = pi.pixBase + (Wp + 1) * CH;   // output interior pixel
#pragma unroll
  for (int j = 0; j < MT; ++j) {
    int mTile = wave + 8 * j;
    if (mTile >= Mtiles) continue;
    int mb = mTile * 16 + kSel * 8;
    union { _Float16 h[8]; uint4 u; } res;
#pragma unroll
    for (int r = 0; r < 8; ++r) {
      float v = acc[j][r] + bias[mb + r];
      res.h[r] = (_Float16)fmaxf(v, 0.0f);
    }
    if (pi.valid) *(uint4*)(outAct + outBase + mb) = res.u;
  }
}

// ---------------- head conv: GEMM + bias -> permuted fp32 output -----------
template <int MT, bool CLS, int H, int W>
__global__ __launch_bounds__(256) void head_wmma(
    const _Float16* __restrict__ act, const _Float16* __restrict__ wgt,
    const float* __restrict__ bias, float* __restrict__ out,
    int rOff, int Mtiles) {
  __shared__ _Float16 ldsB[2][PH_HALFS];
  const int lane = threadIdx.x & 31;
  const int wave = threadIdx.x >> 5;
  const int col  = lane & 15;
  const int kSel = lane >> 4;
  const int blockPix0 = blockIdx.x * 16;

  PixInfo pi = pix_of<H, W>(blockPix0, col);

  v8f zero = {};
  v8f acc[MT];
#pragma unroll
  for (int j = 0; j < MT; ++j) acc[j] = zero;

  gemm_pipeline<MT, H, W>(acc, ldsB, act, wgt, blockPix0, wave, col, kSel, Mtiles);

  // out layout: out[(n*RTOT + rOff + p*A + a)*84 + (CLS ? k : 80 + j)]
#pragma unroll
  for (int j = 0; j < MT; ++j) {
    int mTile = wave + 8 * j;
    if (mTile >= Mtiles) continue;
    if (!pi.valid) continue;
    int mb = mTile * 16 + kSel * 8;      // 8 consecutive out channels
    long rowBase = (long)pi.n * RTOT + rOff + (long)pi.p * AANCH;
    if (CLS) {
      int a  = mb / KCLS;                // 8-block never crosses a boundary (80%8==0)
      int kk = mb - a * KCLS;
      float* o = out + (rowBase + a) * 84 + kk;
      float4 lo, hi;
      lo.x = acc[j][0] + bias[mb + 0]; lo.y = acc[j][1] + bias[mb + 1];
      lo.z = acc[j][2] + bias[mb + 2]; lo.w = acc[j][3] + bias[mb + 3];
      hi.x = acc[j][4] + bias[mb + 4]; hi.y = acc[j][5] + bias[mb + 5];
      hi.z = acc[j][6] + bias[mb + 6]; hi.w = acc[j][7] + bias[mb + 7];
      *(float4*)o       = lo;
      *(float4*)(o + 4) = hi;
    } else {
      int a0 = mb >> 2;                  // box: 4 regressors per anchor
      if (mb < 36) {
        float4 v;
        v.x = acc[j][0] + bias[mb + 0]; v.y = acc[j][1] + bias[mb + 1];
        v.z = acc[j][2] + bias[mb + 2]; v.w = acc[j][3] + bias[mb + 3];
        *(float4*)(out + (rowBase + a0) * 84 + 80) = v;
      }
      if (mb + 4 < 36) {
        float4 v;
        v.x = acc[j][4] + bias[mb + 4]; v.y = acc[j][5] + bias[mb + 5];
        v.z = acc[j][6] + bias[mb + 6]; v.w = acc[j][7] + bias[mb + 7];
        *(float4*)(out + (rowBase + a0 + 1) * 84 + 80) = v;
      }
    }
  }
}

// ---------------------------------------------------------------------------
struct LevelArgs {
  const float *feat;
  const float *clsB[4], *boxB[4];
  const float *cls_score_b, *box_pred_b;
  _Float16 *actF, *actA, *actB, *wT, *wCls, *wBox;
  float* out;
  int rOff;
  long WTSZ;
};

template <int H, int W>
static void run_level(const LevelArgs& a, hipStream_t stream) {
  constexpr int padElems = 2 * (H + 2) * (W + 2) * CH;
  to_padded<H, W><<<(padElems + 255) / 256, 256, 0, stream>>>(a.feat, a.actF);
  constexpr int n8 = padElems / 8;
  zero_f16<<<(n8 + 255) / 256, 256, 0, stream>>>(a.actA, n8);
  zero_f16<<<(n8 + 255) / 256, 256, 0, stream>>>(a.actB, n8);

  constexpr int PT = (2 * H * W + 15) / 16;
  dim3 grid(PT), blk(256);

  // cls tower + cls head
  conv_wmma<2, H, W><<<grid, blk, 0, stream>>>(a.actF, a.wT + 0 * a.WTSZ, a.clsB[0], a.actA, 16);
  conv_wmma<2, H, W><<<grid, blk, 0, stream>>>(a.actA, a.wT + 1 * a.WTSZ, a.clsB[1], a.actB, 16);
  conv_wmma<2, H, W><<<grid, blk, 0, stream>>>(a.actB, a.wT + 2 * a.WTSZ, a.clsB[2], a.actA, 16);
  conv_wmma<2, H, W><<<grid, blk, 0, stream>>>(a.actA, a.wT + 3 * a.WTSZ, a.clsB[3], a.actB, 16);
  head_wmma<6, true, H, W><<<grid, blk, 0, stream>>>(a.actB, a.wCls, a.cls_score_b,
                                                     a.out, a.rOff, 45);
  // box tower + box head
  conv_wmma<2, H, W><<<grid, blk, 0, stream>>>(a.actF, a.wT + 4 * a.WTSZ, a.boxB[0], a.actA, 16);
  conv_wmma<2, H, W><<<grid, blk, 0, stream>>>(a.actA, a.wT + 5 * a.WTSZ, a.boxB[1], a.actB, 16);
  conv_wmma<2, H, W><<<grid, blk, 0, stream>>>(a.actB, a.wT + 6 * a.WTSZ, a.boxB[2], a.actA, 16);
  conv_wmma<2, H, W><<<grid, blk, 0, stream>>>(a.actA, a.wT + 7 * a.WTSZ, a.boxB[3], a.actB, 16);
  head_wmma<1, false, H, W><<<grid, blk, 0, stream>>>(a.actB, a.wBox, a.box_pred_b,
                                                      a.out, a.rOff, 3);
}

extern "C" void kernel_launch(void* const* d_in, const int* in_sizes, int n_in,
                              void* d_out, int out_size, void* d_ws, size_t ws_size,
                              hipStream_t stream) {
  (void)in_sizes; (void)n_in; (void)out_size; (void)ws_size;

  LevelArgs a;
  a.feat = nullptr;
  const float* f[5];
  for (int i = 0; i < 5; ++i) f[i] = (const float*)d_in[i];
  const float *clsW[4], *boxW[4];
  for (int i = 0; i < 4; ++i) {
    clsW[i]   = (const float*)d_in[5 + 2 * i];
    a.clsB[i] = (const float*)d_in[6 + 2 * i];
    boxW[i]   = (const float*)d_in[13 + 2 * i];
    a.boxB[i] = (const float*)d_in[14 + 2 * i];
  }
  const float* cls_score_w = (const float*)d_in[21];
  a.cls_score_b            = (const float*)d_in[22];
  const float* box_pred_w  = (const float*)d_in[23];
  a.box_pred_b             = (const float*)d_in[24];
  a.out = (float*)d_out;

  // workspace layout (halfs): 3 padded act buffers + packed weights (~45 MB)
  const long ACTSZ = (long)2 * 102 * 102 * 256;   // 5,326,848 halfs
  a.WTSZ = (long)CH * KDIM;                       //   589,824 halfs
  _Float16* ws = (_Float16*)d_ws;
  a.actF = ws;
  a.actA = ws + ACTSZ;
  a.actB = ws + 2 * ACTSZ;
  a.wT   = ws + 3 * ACTSZ;                        // 8 tower mats
  a.wCls = a.wT + 8 * a.WTSZ;                     // 720 x 2304
  a.wBox = a.wCls + (long)720 * KDIM;             // 48  x 2304 (zero-padded)

  // ---- repack weights to f16 A-matrices ----
  {
    int tot = CH * KDIM, g = (tot + 255) / 256;
    for (int k = 0; k < 4; ++k)
      pack_w<<<g, 256, 0, stream>>>(clsW[k], a.wT + k * a.WTSZ, CH, CH);
    for (int k = 0; k < 4; ++k)
      pack_w<<<g, 256, 0, stream>>>(boxW[k], a.wT + (4 + k) * a.WTSZ, CH, CH);
    int totC = 720 * KDIM;
    pack_w<<<(totC + 255) / 256, 256, 0, stream>>>(cls_score_w, a.wCls, 720, 720);
    int totB = 48 * KDIM;
    pack_w<<<(totB + 255) / 256, 256, 0, stream>>>(box_pred_w, a.wBox, 36, 48);
  }

  a.feat = f[0]; a.rOff = 0;      run_level<100, 100>(a, stream);
  a.feat = f[1]; a.rOff = 90000;  run_level<50, 50>(a, stream);
  a.feat = f[2]; a.rOff = 112500; run_level<25, 25>(a, stream);
  a.feat = f[3]; a.rOff = 118125; run_level<13, 13>(a, stream);
  a.feat = f[4]; a.rOff = 119646; run_level<7, 7>(a, stream);
}
